// FastFourierConvolution_35897336660138
// MI455X (gfx1250) — matrix operand order repaired and checked by reading, not compiled, not run on hardware
//
#include <hip/hip_runtime.h>
#include <hip/hip_bf16.h>
#include <math.h>

typedef __attribute__((ext_vector_type(16))) __bf16 v16bf;
typedef __attribute__((ext_vector_type(8)))  float  v8f;

#define FFC_EPS 1e-5f

// ---------------------------------------------------------------------------
// 1x1 conv (channel-mixing GEMM) via v_wmma_f32_16x16x32_bf16, templated on
// (COUT, CIN, RESID) so all tile math is compile-time and the epilogue is
// branch-free.
//   out[b, ooff+co, p] = sum_ci w[co,ci] * in[b, ioff+ci, p] + bias[co]
//                        (+ resid[b, roff+co, p] if RESID)
// A (weights) and B (pixels) are staged into LDS pre-swizzled into WMMA
// fragment order: each lane's 16 bf16 operand elements are 32 contiguous
// bytes -> operand load = 2x ds_load_b128.
//
// Fragment maps (ISA 7.12.2, wave32):
//  A 16x32 bf16: lane<16: M=lane, K in {0..7,16..23}; lane>=16: K in {8..15,24..31}
//     inverse: lg=(kr>>3)&1, i=(kr&7)+((kr&16)?8:0)
//  B 32x16 bf16: lane L: N=L&15; K as for A: L<16 {0..7,16..23}, L>=16 {8..15,24..31}
//     inverse: lg=(kr>>3)&1, i=(kr&7)+((kr&16)?8:0)
//  C/D f32:      lane L: N=L&15; vgpr r: M = r + (L>=16?8:0)
// ---------------------------------------------------------------------------
template<int COUT, int CIN, bool RESID>
__global__ __launch_bounds__(256) void ffc_conv1x1_wmma(
    float* __restrict__ out, int ostr, int ooff,
    const float* __restrict__ in, int istr, int ioff,
    const float* __restrict__ w, const float* __restrict__ bias,
    const float* __restrict__ resid, int rstr, int roff, int P)
{
  constexpr int wavesM = COUT / 16;          // 2 or 4
  constexpr int wavesN = 8 / wavesM;         // 4 or 2
  constexpr int NPB    = wavesN * 16;        // 64 or 32 pixels per block
  constexpr int KB     = CIN / 32;           // 1 or 2 k-steps

  __shared__ alignas(32) __bf16 w_s[KB * COUT * 32];  // fragment-swizzled A
  __shared__ alignas(32) __bf16 x_s[KB * NPB  * 32];  // fragment-swizzled B
  __shared__ float bias_s[COUT];

  const int tid = threadIdx.x;

  // ---- stage weights (COUT x CIN row-major) swizzled to A-fragment order
  for (int i = tid; i < COUT * CIN; i += 256) {
    const int m  = i / CIN;
    const int k  = i - m * CIN;
    const int kb = k >> 5, kr = k & 31;
    const int lg = (kr >> 3) & 1;
    const int ii = (kr & 7) + ((kr & 16) ? 8 : 0);
    w_s[(((kb * COUT + m) << 1) + lg) * 16 + ii] = (__bf16)w[i];
  }
  if (tid < COUT) bias_s[tid] = bias[tid];

  const int p0 = blockIdx.x * NPB;           // pixel index over B*P
  const int b  = p0 / P;
  const int pp = p0 - b * P;

  // ---- stage input tile [CIN][NPB] swizzled to B-fragment order
  constexpr int TOT4 = (CIN * NPB) / 4;
  for (int i = tid; i < TOT4; i += 256) {
    const int i4 = i << 2;
    const int ci = i4 / NPB;                 // compile-time shift
    const int n  = i4 & (NPB - 1);
    const float4 v4 = *(const float4*)
        &in[((size_t)(b * istr + ioff + ci)) * (size_t)P + pp + n];
    const int kb = ci >> 5, kr = ci & 31;
    const int lg = (kr >> 3) & 1, ii = (kr & 7) + ((kr & 16) ? 8 : 0);
    x_s[(((kb * NPB + n + 0) << 1) + lg) * 16 + ii] = (__bf16)v4.x;
    x_s[(((kb * NPB + n + 1) << 1) + lg) * 16 + ii] = (__bf16)v4.y;
    x_s[(((kb * NPB + n + 2) << 1) + lg) * 16 + ii] = (__bf16)v4.z;
    x_s[(((kb * NPB + n + 3) << 1) + lg) * 16 + ii] = (__bf16)v4.w;
  }
  __syncthreads();

  const int wave = tid >> 5, lane = tid & 31;
  const int mtile = wave % wavesM;
  const int ntile = wave / wavesM;
  const int m0 = mtile * 16, n0 = ntile * 16;
  const int lg = lane >> 4,  ml = lane & 15;

  v8f acc = {0.f, 0.f, 0.f, 0.f, 0.f, 0.f, 0.f, 0.f};
#pragma unroll
  for (int kb = 0; kb < KB; kb++) {
    const v16bf afrag =
        *(const v16bf*)(w_s + (((kb * COUT + m0 + ml) << 1) + lg) * 16);
    const v16bf bfrag =
        *(const v16bf*)(x_s + (((kb * NPB + n0 + ml) << 1) + lg) * 16);
    acc = __builtin_amdgcn_wmma_f32_16x16x32_bf16(
        false, afrag, false, bfrag, (short)0, acc, false, false);
  }

  const int mrow = m0 + lg * 8;
  const size_t obase =
      ((size_t)(b * ostr + ooff + mrow)) * (size_t)P + pp + n0 + ml;
  if (RESID) {
    const size_t rbase =
        ((size_t)(b * rstr + roff + mrow)) * (size_t)P + pp + n0 + ml;
#pragma unroll
    for (int r = 0; r < 8; r++) {
      out[obase + (size_t)r * (size_t)P] =
          acc[r] + bias_s[mrow + r] + resid[rbase + (size_t)r * (size_t)P];
    }
  } else {
#pragma unroll
    for (int r = 0; r < 8; r++) {
      out[obase + (size_t)r * (size_t)P] = acc[r] + bias_s[mrow + r];
    }
  }
}

template<int COUT, int CIN>
static void launch_conv(float* o, int os, int oo, const float* in, int is, int io,
                        const float* w, const float* bi,
                        const float* r, int rs, int ro, int P, hipStream_t stream)
{
  constexpr int NPB = (8 / (COUT / 16)) * 16;
  const dim3 grid((4 * P) / NPB), blk(256);
  if (r)
    ffc_conv1x1_wmma<COUT, CIN, true><<<grid, blk, 0, stream>>>(
        o, os, oo, in, is, io, w, bi, r, rs, ro, P);
  else
    ffc_conv1x1_wmma<COUT, CIN, false><<<grid, blk, 0, stream>>>(
        o, os, oo, in, is, io, w, bi, r, rs, ro, P);
}

// ---------------------------------------------------------------------------
// Wave-level radix-2 DIF FFT in LDS.  Natural-order input, bit-reversed
// output (callers index results with rev()).  All global memory traffic in
// the FFT kernels is natural-order/coalesced; bit reversal happens only on
// LDS-side indexing.  sign = -1 forward, +1 inverse (unnormalized).
// ---------------------------------------------------------------------------
__device__ __forceinline__ int ffc_rev(int j, int logN) {
  return (int)(__brev((unsigned)j) >> (32 - logN));
}

__device__ __forceinline__ void ffc_wave_fft_dif(float2* buf, int N, int logN,
                                                 int lane, float sign)
{
  for (int s = logN; s >= 1; s--) {
    const int half  = 1 << (s - 1);
    const float wst = sign * 6.28318530717958647692f / (float)(1 << s);
    for (int i = lane; i < (N >> 1); i += 32) {
      const int j   = i & (half - 1);
      const int blk = (i >> (s - 1)) << s;
      float2 u = buf[blk + j];
      float2 t = buf[blk + j + half];
      float2 sm; sm.x = u.x + t.x; sm.y = u.y + t.y;
      float2 df; df.x = u.x - t.x; df.y = u.y - t.y;
      float sn, cs;
      __sincosf(wst * (float)j, &sn, &cs);
      float2 o; o.x = df.x * cs - df.y * sn; o.y = df.x * sn + df.y * cs;
      buf[blk + j]        = sm;
      buf[blk + j + half] = o;
    }
    __syncthreads();
  }
}

// Forward row FFT of real input -> half spectrum Z (B,C,H,Wf) complex.
// quad != 0: gather quadrant-packed input from channels ioff..ioff+7 of a
// (B,istr,256,256) tensor, 128x128 sub-tiles.
__global__ __launch_bounds__(256) void ffc_fft_rows_fwd(
    const float* __restrict__ src, int istr, int ioff,
    int C, int H, int W, int logW, int Wf, int quad,
    float2* __restrict__ Z)
{
  __shared__ float2 fbuf[8 * 256];
  const int wave = threadIdx.x >> 5, lane = threadIdx.x & 31;
  float2* buf = fbuf + wave * 256;

  const int row = blockIdx.x * 8 + wave;
  const int h = row % H;
  const int t = row / H;
  const int c = t % C;
  const int b = t / C;

  for (int j = lane; j < W; j += 32) {
    float xv;
    if (quad) {
      const int q  = c >> 3;
      const int cc = ioff + (c & 7);
      const int hs = h + ((q >> 1) ? 128 : 0);
      const int ws = j + ((q & 1)  ? 128 : 0);
      xv = src[((size_t)(b * istr + cc)) * 65536u + (size_t)hs * 256 + ws];
    } else {
      xv = src[(((size_t)(b * istr + ioff + c)) * H + h) * (size_t)W + j];
    }
    float2 v; v.x = xv; v.y = 0.f;
    buf[j] = v;
  }
  __syncthreads();
  ffc_wave_fft_dif(buf, W, logW, lane, -1.0f);
  for (int v = lane; v < Wf; v += 32)
    Z[(((size_t)(b * C + c)) * H + h) * (size_t)Wf + v] = buf[ffc_rev(v, logW)];
}

// Forward column FFT of Z -> yc real tensor (B,2C,H,Wf): Re at ch c, Im at C+c.
__global__ __launch_bounds__(256) void ffc_fft_cols_fwd(
    const float2* __restrict__ Z, float* __restrict__ yc,
    int C, int H, int logH, int Wf)
{
  __shared__ float2 fbuf[8 * 256];
  const int wave = threadIdx.x >> 5, lane = threadIdx.x & 31;
  float2* buf = fbuf + wave * 256;

  const int col = blockIdx.x * 8 + wave;
  const int v = col % Wf;
  const int t = col / Wf;
  const int c = t % C;
  const int b = t / C;

  const float2* zc = Z + ((size_t)(b * C + c)) * H * (size_t)Wf + v;
  for (int j = lane; j < H; j += 32) buf[j] = zc[(size_t)j * Wf];
  __syncthreads();
  ffc_wave_fft_dif(buf, H, logH, lane, -1.0f);
  const int C2 = 2 * C;
  for (int u = lane; u < H; u += 32) {
    const float2 val = buf[ffc_rev(u, logH)];
    yc[(((size_t)(b * C2 + c))     * H + u) * (size_t)Wf + v] = val.x;
    yc[(((size_t)(b * C2 + C + c)) * H + u) * (size_t)Wf + v] = val.y;
  }
}

// Inverse column FFT (sign +1, unnormalized) of h1 -> G1, with the BatchNorm
// + ReLU folded into the loader:  re channel c uses (mu,var,gamma,beta)[c],
// im channel C+c uses [C+c].  stats = {sum[0..C2), sumsq[0..C2)}.
__global__ __launch_bounds__(256) void ffc_ifft_cols_bn(
    const float* __restrict__ yc2, float2* __restrict__ G1,
    const float* __restrict__ stats,
    const float* __restrict__ gamma, const float* __restrict__ beta,
    int C, int H, int logH, int Wf, float invN)
{
  __shared__ float2 fbuf[8 * 256];
  const int wave = threadIdx.x >> 5, lane = threadIdx.x & 31;
  float2* buf = fbuf + wave * 256;

  const int col = blockIdx.x * 8 + wave;
  const int v = col % Wf;
  const int t = col / Wf;
  const int c = t % C;
  const int b = t / C;

  const int C2 = 2 * C;
  // per-wave normalization constants for the two source channels
  const float mu_r  = stats[c] * invN;
  const float var_r = stats[C2 + c] * invN - mu_r * mu_r;
  const float sc_r  = rsqrtf(var_r + FFC_EPS) * gamma[c];
  const float sh_r  = beta[c] - mu_r * sc_r;
  const float mu_i  = stats[C + c] * invN;
  const float var_i = stats[C2 + C + c] * invN - mu_i * mu_i;
  const float sc_i  = rsqrtf(var_i + FFC_EPS) * gamma[C + c];
  const float sh_i  = beta[C + c] - mu_i * sc_i;

  const float* re = yc2 + ((size_t)(b * C2 + c))     * H * (size_t)Wf + v;
  const float* im = yc2 + ((size_t)(b * C2 + C + c)) * H * (size_t)Wf + v;
  for (int j = lane; j < H; j += 32) {
    float xr = re[(size_t)j * Wf] * sc_r + sh_r;
    float xi = im[(size_t)j * Wf] * sc_i + sh_i;
    float2 t2;
    t2.x = xr > 0.f ? xr : 0.f;
    t2.y = xi > 0.f ? xi : 0.f;
    buf[j] = t2;
  }
  __syncthreads();
  ffc_wave_fft_dif(buf, H, logH, lane, 1.0f);
  for (int u = lane; u < H; u += 32)
    G1[(((size_t)(b * C + c)) * H + u) * (size_t)Wf + v] = buf[ffc_rev(u, logH)];
}

// Inverse row FFT with Hermitian reconstruction; writes real part * scale.
//   r[0] = conj(G1[0]); r[1..Wf-1] = G1[v]; r[v>=Wf] = conj(G1[W-v])
__global__ __launch_bounds__(256) void ffc_ifft_rows_final(
    const float2* __restrict__ G1, float* __restrict__ dst,
    int C, int H, int W, int logW, int Wf, float scale)
{
  __shared__ float2 fbuf[8 * 256];
  const int wave = threadIdx.x >> 5, lane = threadIdx.x & 31;
  float2* buf = fbuf + wave * 256;

  const int row = blockIdx.x * 8 + wave;
  const int y = row % H;
  const int t = row / H;
  const int c = t % C;
  const int b = t / C;

  const float2* g = G1 + (((size_t)(b * C + c)) * H + y) * (size_t)Wf;
  for (int j = lane; j < W; j += 32) {
    float2 val;
    if (j == 0)           { float2 a = g[0];     val.x = a.x; val.y = -a.y; }
    else if (j <= Wf - 1) { val = g[j]; }
    else                  { float2 a = g[W - j]; val.x = a.x; val.y = -a.y; }
    buf[j] = val;
  }
  __syncthreads();
  ffc_wave_fft_dif(buf, W, logW, lane, 1.0f);
  for (int x = lane; x < W; x += 32)
    dst[(((size_t)(b * C + c)) * H + y) * (size_t)W + x] =
        buf[ffc_rev(x, logW)].x * scale;
}

// ---------------------------------------------------------------------------
// BatchNorm statistics: per-channel sum / sumsq via LDS tree + global atomics.
// ---------------------------------------------------------------------------
__global__ void ffc_zero(float* __restrict__ p, int n)
{
  const int i = blockIdx.x * blockDim.x + threadIdx.x;
  if (i < n) p[i] = 0.f;
}

__global__ __launch_bounds__(256) void ffc_reduce_stats(
    const float* __restrict__ x, int C2, int P, int B, float* __restrict__ stats)
{
  __shared__ float ssum[256];
  __shared__ float ssq[256];
  const int c  = blockIdx.x;
  const int P4 = P >> 2;
  const long total4 = (long)B * P4;
  float s = 0.f, q = 0.f;
  for (long i = (long)blockIdx.y * 256 + threadIdx.x; i < total4;
       i += (long)gridDim.y * 256) {
    const int b  = (int)(i / P4);
    const int p4 = (int)(i - (long)b * P4);
    const float4 v = *(const float4*)
        &x[((size_t)(b * C2 + c)) * (size_t)P + (size_t)p4 * 4];
    s += v.x + v.y + v.z + v.w;
    q += v.x * v.x + v.y * v.y + v.z * v.z + v.w * v.w;
  }
  ssum[threadIdx.x] = s; ssq[threadIdx.x] = q;
  __syncthreads();
  for (int off = 128; off > 0; off >>= 1) {
    if ((int)threadIdx.x < off) {
      ssum[threadIdx.x] += ssum[threadIdx.x + off];
      ssq[threadIdx.x]  += ssq[threadIdx.x + off];
    }
    __syncthreads();
  }
  if (threadIdx.x == 0) {
    atomicAdd(&stats[c], ssum[0]);
    atomicAdd(&stats[C2 + c], ssq[0]);
  }
}

// ysum = y_g + 2x2-tile(local): y_g (B,32,256,256) += local[(b,c), y&127, x&127]
// float4: 4-aligned x never crosses the &127 wrap.
__global__ __launch_bounds__(256) void ffc_combine(
    float* __restrict__ yg, const float* __restrict__ loc)
{
  const size_t i4  = ((size_t)blockIdx.x * 256 + threadIdx.x) << 2;
  const int    x   = (int)(i4 & 255);
  const int    y   = (int)((i4 >> 8) & 255);
  const size_t bc  = i4 >> 16;                 // b*32 + c
  const float4 a = *(const float4*)&yg[i4];
  const float4 l = *(const float4*)
      &loc[(bc * 128 + (size_t)(y & 127)) * 128 + (size_t)(x & 127)];
  float4 r; r.x = a.x + l.x; r.y = a.y + l.y; r.z = a.z + l.z; r.w = a.w + l.w;
  *(float4*)&yg[i4] = r;
}

// ---------------------------------------------------------------------------
extern "C" void kernel_launch(void* const* d_in, const int* in_sizes, int n_in,
                              void* d_out, int out_size, void* d_ws, size_t ws_size,
                              hipStream_t stream) {
  const float* x       = (const float*)d_in[0];
  const float* w_local = (const float*)d_in[1];
  const float* b_local = (const float*)d_in[2];
  const float* w_in    = (const float*)d_in[3];
  const float* b_in    = (const float*)d_in[4];
  const float* w_out   = (const float*)d_in[5];
  const float* b_out   = (const float*)d_in[6];
  const float* w_g2l   = (const float*)d_in[7];
  const float* b_g2l   = (const float*)d_in[8];
  const float* w_l2g   = (const float*)d_in[9];
  const float* b_l2g   = (const float*)d_in[10];
  const float* w_fu1   = (const float*)d_in[11];
  const float* b_fu1   = (const float*)d_in[12];
  const float* g_fu1   = (const float*)d_in[13];
  const float* be_fu1  = (const float*)d_in[14];
  const float* w_fu2   = (const float*)d_in[15];
  const float* b_fu2   = (const float*)d_in[16];
  const float* g_fu2   = (const float*)d_in[17];
  const float* be_fu2  = (const float*)d_in[18];
  float* out = (float*)d_out;

  // ---- workspace layout (floats) ----
  const size_t SPAT32 = (size_t)4 * 32 * 256 * 256;   // 8,388,608
  const size_t FREQ   = (size_t)4 * 64 * 256 * 129;   // 8,454,144
  float* y_in  = (float*)d_ws;          // (4,32,256,256) conv_in output
  float* y_g   = y_in + SPAT32;         // (4,32,256,256) FU1 output / ysum
  float* F0    = y_g  + SPAT32;         // freq scratch A (also float2 view)
  float* F1    = F0   + FREQ;           // freq scratch B
  float* locb  = F1   + FREQ;           // (4,32,128,128) local FU output
  float* stats = locb + (size_t)4 * 32 * 128 * 128;   // 128 floats
  (void)ws_size; (void)in_sizes; (void)n_in; (void)out_size;

  const int B = 4, HW = 65536;
  const int P1 = 256 * 129;   // 33024, FU1 freq pixels/image
  const int P2 = 128 * 65;    // 8320,  FU2 freq pixels/image

  // 1) y_in = conv1x1(x_g, w_in) : (4,32,256,256)
  launch_conv<32, 64>(y_in, 32, 0, x, 128, 0, w_in, b_in,
                      nullptr, 0, 0, HW, stream);

  // 2) Fourier unit 1 on y_in (H=W=256, Wf=129)
  ffc_fft_rows_fwd<<<dim3(4 * 32 * 256 / 8), dim3(256), 0, stream>>>(
      y_in, 32, 0, 32, 256, 256, 8, 129, 0, (float2*)F0);
  ffc_fft_cols_fwd<<<dim3(4 * 32 * 129 / 8), dim3(256), 0, stream>>>(
      (float2*)F0, F1, 32, 256, 8, 129);
  launch_conv<64, 64>(F0, 64, 0, F1, 64, 0, w_fu1, b_fu1,
                      nullptr, 0, 0, P1, stream);
  ffc_zero<<<dim3(1), dim3(128), 0, stream>>>(stats, 128);
  ffc_reduce_stats<<<dim3(64, 256), dim3(256), 0, stream>>>(F0, 64, P1, B, stats);
  ffc_ifft_cols_bn<<<dim3(4 * 32 * 129 / 8), dim3(256), 0, stream>>>(
      F0, (float2*)F1, stats, g_fu1, be_fu1, 32, 256, 8, 129,
      1.0f / (float)(B * P1));
  ffc_ifft_rows_final<<<dim3(4 * 32 * 256 / 8), dim3(256), 0, stream>>>(
      (float2*)F1, y_g, 32, 256, 256, 8, 129, 1.0f / 65536.0f);

  // 3) Local Fourier unit on quadrant-packed y_in[:,24:32] (H=W=128, Wf=65)
  ffc_fft_rows_fwd<<<dim3(4 * 32 * 128 / 8), dim3(256), 0, stream>>>(
      y_in, 32, 24, 32, 128, 128, 7, 65, 1, (float2*)F0);
  ffc_fft_cols_fwd<<<dim3(4 * 32 * 65 / 8), dim3(256), 0, stream>>>(
      (float2*)F0, F1, 32, 128, 7, 65);
  launch_conv<64, 64>(F0, 64, 0, F1, 64, 0, w_fu2, b_fu2,
                      nullptr, 0, 0, P2, stream);
  ffc_zero<<<dim3(1), dim3(128), 0, stream>>>(stats, 128);
  ffc_reduce_stats<<<dim3(64, 64), dim3(256), 0, stream>>>(F0, 64, P2, B, stats);
  ffc_ifft_cols_bn<<<dim3(4 * 32 * 65 / 8), dim3(256), 0, stream>>>(
      F0, (float2*)F1, stats, g_fu2, be_fu2, 32, 128, 7, 65,
      1.0f / (float)(B * P2));
  ffc_ifft_rows_final<<<dim3(4 * 32 * 128 / 8), dim3(256), 0, stream>>>(
      (float2*)F1, locb, 32, 128, 128, 7, 65, 1.0f / 16384.0f);

  // 4) ysum = y_g + tile2x2(local)   (in place into y_g)
  ffc_combine<<<dim3((4 * 32 * 65536) / (256 * 4)), dim3(256), 0, stream>>>(
      y_g, locb);

  // 5) Y_g = conv(ysum, w_out) + conv(x_l, w_l2g)  -> out channels [0,64)
  launch_conv<64, 64>(out, 128, 0, x, 128, 64, w_l2g, b_l2g,
                      nullptr, 0, 0, HW, stream);
  launch_conv<64, 32>(out, 128, 0, y_g, 32, 0, w_out, b_out,
                      out, 128, 0, HW, stream);

  // 6) Y_l = conv(x_l, w_local) + conv(x_g, w_g2l) -> out channels [64,128)
  launch_conv<64, 64>(out, 128, 64, x, 128, 64, w_local, b_local,
                      nullptr, 0, 0, HW, stream);
  launch_conv<64, 64>(out, 128, 64, x, 128, 0, w_g2l, b_g2l,
                      out, 128, 64, HW, stream);
}
